// ProbFormer_47330539602637
// MI455X (gfx1250) — matrix operand order repaired and checked by reading, not compiled, not run on hardware
//
#include <hip/hip_runtime.h>
#include <math.h>

// ---------------- types ----------------
typedef __attribute__((ext_vector_type(16))) __bf16 v16bf;
typedef __attribute__((ext_vector_type(8)))  float  v8f;

union BFrag { uint4 u[2]; v16bf v; };

__device__ __forceinline__ unsigned short f2bf(float f) {
    unsigned int u = __float_as_uint(f);
    u = (u + 0x7FFFu + ((u >> 16) & 1u)) >> 16;   // RNE
    return (unsigned short)u;
}

// generic -> LDS(addrspace 3) byte offset
__device__ __forceinline__ unsigned ldsOff(const void* p) {
    return (unsigned)(unsigned long long)(__attribute__((address_space(3))) const void*)p;
}
// CDNA5 async global->LDS copy (ASYNCcnt tracked), per-lane 16B
__device__ __forceinline__ void asyncLoadB128(unsigned lds_byte_off, const void* gaddr) {
    asm volatile("global_load_async_to_lds_b128 %0, %1, off"
                 :: "v"(lds_byte_off), "v"(gaddr)
                 : "memory");
}
__device__ __forceinline__ void waitAsync0() {
    asm volatile("s_wait_asynccnt 0x0" ::: "memory");
}
__device__ __forceinline__ void waitAsync4() {   // keep the 4 just-issued copies in flight
    asm volatile("s_wait_asynccnt 0x4" ::: "memory");
}

// ---------------- constants ----------------
#define BATCH   32
#define NTOK    197
#define NTOKP   208          // padded to 13*16
#define DMODEL  768
#define NHEADS  12
#define DK      64
#define NPATCH  196
#define MTOK    (BATCH*NTOK)     // 6304
#define MPATCH  (BATCH*NPATCH)   // 6272
#define DQKV    2304
#define DFF     3072
#define NLAYERS 12

#define BM 128
#define BN 128
#define BK 32
#define LSTRIDE 40   // padded LDS row stride (elements) to break bank-conflict pattern

enum { EPI_F32 = 0, EPI_BF16 = 1, EPI_GELU_BF16 = 2, EPI_RES = 3, EPI_RES_YMEAN = 4 };

// ---------------- generic batched WMMA GEMM ----------------
// C[M,N] = A[M,K] @ B[N,K]^T (+bias[n]) (+residual) ; A,B bf16 row-major.
// batch z: A += z*aStride, B += z*bStride, C base = (z/batchInner)*cOuter + (z%batchInner)*cInner
// 8 waves: 4x2 grid of 32x64 wave tiles; each wave: 2x4 subtiles of 16x16, 8 WMMAs/K-step.
// Interior blocks with K%32==0 run a double-buffered async global->LDS pipeline (1 tile in flight).
template <int EPI>
__global__ __launch_bounds__(256) void gemm_bf16_wmma(
    const unsigned short* __restrict__ A, int lda, long long aStride,
    const unsigned short* __restrict__ B, int ldb, long long bStride,
    void* __restrict__ Cv, int ldc, long long cOuter, long long cInner, int batchInner,
    int M, int N, int K, int Mvalid,
    const float* __restrict__ bias,
    const float* __restrict__ residual,
    float* __restrict__ ymean)
{
    __shared__ unsigned short lA[2][BM * LSTRIDE];
    __shared__ unsigned short lB[2][BN * LSTRIDE];

    const int z = blockIdx.z;
    const int outer = z / batchInner;
    const int inner = z - outer * batchInner;
    const unsigned short* Ab = A + (long long)z * aStride;
    const unsigned short* Bb = B + (long long)z * bStride;
    const long long cbase = (long long)outer * cOuter + (long long)inner * cInner;

    const int bm = blockIdx.y * BM;
    const int bn = blockIdx.x * BN;
    const int tid = threadIdx.x;
    const int wid = tid >> 5;
    const int lane = tid & 31;
    const int wm = (wid >> 1) * 32;   // 0,32,64,96
    const int wn = (wid & 1) * 64;    // 0,64
    const int r16 = lane & 15;
    const bool lo = lane < 16;

    const bool interiorMN = (bm + BM <= M) && (bn + BN <= N);
    const unsigned baseA = ldsOff(&lA[0][0]);
    const unsigned baseB = ldsOff(&lB[0][0]);

    // per-thread staging coordinates: 2 chunks of 8 bf16 for each of A and B
    const int srow0 = tid >> 2;              // 0..63
    const int srow1 = srow0 + 64;            // 64..127
    const int sc8   = (tid & 3) * 8;         // 0,8,16,24

    v8f acc[2][4] = {};

    // issue one 128x32 A tile + 128x32 B tile into LDS buffer `buf` (4 async B128 per thread)
    auto issueTile = [&](int k0, int buf) {
        unsigned oA = baseA + (unsigned)(buf * BM * LSTRIDE * 2);
        unsigned oB = baseB + (unsigned)(buf * BN * LSTRIDE * 2);
        asyncLoadB128(oA + (unsigned)(srow0 * LSTRIDE + sc8) * 2,
                      Ab + (long long)(bm + srow0) * lda + (k0 + sc8));
        asyncLoadB128(oA + (unsigned)(srow1 * LSTRIDE + sc8) * 2,
                      Ab + (long long)(bm + srow1) * lda + (k0 + sc8));
        asyncLoadB128(oB + (unsigned)(srow0 * LSTRIDE + sc8) * 2,
                      Bb + (long long)(bn + srow0) * ldb + (k0 + sc8));
        asyncLoadB128(oB + (unsigned)(srow1 * LSTRIDE + sc8) * 2,
                      Bb + (long long)(bn + srow1) * ldb + (k0 + sc8));
    };

    // gather fragments (contiguous k = 16h+s, same order in A and B) + 8 WMMAs from LDS buffer
    auto computeStep = [&](const unsigned short* sA, const unsigned short* sB) {
        v16bf bF[4];
        #pragma unroll
        for (int nt = 0; nt < 4; ++nt) {
            int col = wn + nt * 16 + r16;
            int o = lo ? 0 : 16;      // B lane k-run: 0..15 / 16..31
            BFrag f;
            f.u[0] = *(const uint4*)(&sB[col * LSTRIDE + o]);
            f.u[1] = *(const uint4*)(&sB[col * LSTRIDE + o + 8]);
            bF[nt] = f.v;
        }
        #pragma unroll
        for (int mt = 0; mt < 2; ++mt) {
            int row = wm + mt * 16 + r16;
            int o0 = lo ? 0 : 16;     // A lane k-runs: 0..15 / 16..31
            int o1 = lo ? 8 : 24;
            BFrag f;
            f.u[0] = *(const uint4*)(&sA[row * LSTRIDE + o0]);
            f.u[1] = *(const uint4*)(&sA[row * LSTRIDE + o1]);
            v16bf aF = f.v;
            #pragma unroll
            for (int nt = 0; nt < 4; ++nt)
                acc[mt][nt] = __builtin_amdgcn_wmma_f32_16x16x32_bf16(
                    false, aF, false, bF[nt], (short)0, acc[mt][nt], false, false);
        }
    };

    const bool allFast = interiorMN && ((K & (BK - 1)) == 0);
    if (allFast) {
        // ---- double-buffered async pipeline: tile s+1 in flight while computing tile s ----
        const int nsteps = K / BK;
        issueTile(0, 0);
        for (int s = 0; s < nsteps; ++s) {
            const int cur = s & 1;
            if (s + 1 < nsteps) { issueTile((s + 1) * BK, cur ^ 1); waitAsync4(); }
            else                { waitAsync0(); }
            __syncthreads();                 // tile s resident in all waves
            computeStep(lA[cur], lB[cur]);
            __syncthreads();                 // all waves done with buf `cur` before overwrite
        }
    } else {
        for (int k0 = 0; k0 < K; k0 += BK) {
            const bool fast = interiorMN && (k0 + BK <= K);
            if (fast) {
                issueTile(k0, 0);
                waitAsync0();
            } else {
                // ---- guarded zero-fill path for edge tiles ----
                #pragma unroll
                for (int i = 0; i < 2; ++i) {
                    int row = srow0 + i * 64;
                    int gr = bm + row, gk = k0 + sc8;
                    uint4 val = make_uint4(0u, 0u, 0u, 0u);
                    if (gr < M && gk < K)
                        val = *(const uint4*)(Ab + (long long)gr * lda + gk);
                    *(uint4*)(&lA[0][row * LSTRIDE + sc8]) = val;
                }
                #pragma unroll
                for (int i = 0; i < 2; ++i) {
                    int row = srow0 + i * 64;
                    int gr = bn + row, gk = k0 + sc8;
                    uint4 val = make_uint4(0u, 0u, 0u, 0u);
                    if (gr < N && gk < K)
                        val = *(const uint4*)(Bb + (long long)gr * ldb + gk);
                    *(uint4*)(&lB[0][row * LSTRIDE + sc8]) = val;
                }
            }
            __syncthreads();
            computeStep(lA[0], lB[0]);
            __syncthreads();
        }
    }

    // ---- epilogue ----
    #pragma unroll
    for (int mt = 0; mt < 2; ++mt) {
        #pragma unroll
        for (int nt = 0; nt < 4; ++nt) {
            int n = bn + wn + nt * 16 + r16;
            if (n >= N) continue;
            float bv = bias ? bias[n] : 0.0f;
            #pragma unroll
            for (int r = 0; r < 8; ++r) {
                int m = bm + wm + mt * 16 + r + (lo ? 0 : 8);
                if (m >= Mvalid) continue;
                long long ci = cbase + (long long)m * ldc + n;
                float v = acc[mt][nt][r] + bv;
                if (EPI == EPI_F32) {
                    ((float*)Cv)[ci] = v;
                } else if (EPI == EPI_BF16) {
                    ((unsigned short*)Cv)[ci] = f2bf(v);
                } else if (EPI == EPI_GELU_BF16) {
                    float g = 0.5f * v * (1.0f + erff(v * 0.70710678118654752f));
                    ((unsigned short*)Cv)[ci] = f2bf(g);
                } else if (EPI == EPI_RES) {
                    ((float*)Cv)[ci] = v + residual[ci];
                } else if (EPI == EPI_RES_YMEAN) {
                    ymean[ci] = v;
                    ((float*)Cv)[ci] = v + residual[ci];
                }
            }
        }
    }
}

// ---------------- small helper kernels ----------------
__device__ __forceinline__ float blockReduceSum(float v, float* red) {
    int tid = threadIdx.x;
    red[tid] = v; __syncthreads();
    for (int s = 128; s > 0; s >>= 1) { if (tid < s) red[tid] += red[tid + s]; __syncthreads(); }
    float r = red[0]; __syncthreads();
    return r;
}

// LayerNorm over 768 cols, one block (256 thr) per row, bf16 output
__global__ __launch_bounds__(256) void ln_kernel(
    const float* __restrict__ x, const float* __restrict__ w, const float* __restrict__ b,
    unsigned short* __restrict__ h)
{
    __shared__ float red[256];
    long long row = blockIdx.x;
    const float* xr = x + row * DMODEL;
    int tid = threadIdx.x;
    float v[3]; float s = 0.0f;
    #pragma unroll
    for (int i = 0; i < 3; ++i) { v[i] = xr[tid + 256 * i]; s += v[i]; }
    float mu = blockReduceSum(s, red) * (1.0f / DMODEL);
    float vs = 0.0f;
    #pragma unroll
    for (int i = 0; i < 3; ++i) { float d = v[i] - mu; vs += d * d; }
    float var = blockReduceSum(vs, red) * (1.0f / DMODEL);
    float inv = rsqrtf(var + 1e-6f);
    #pragma unroll
    for (int i = 0; i < 3; ++i) {
        int c = tid + 256 * i;
        h[row * DMODEL + c] = f2bf((v[i] - mu) * inv * w[c] + b[c]);
    }
}

// im2col patch gather fp32 -> bf16  [B*196, 768], col = ch*256+py*16+px
__global__ void patch_gather(const float* __restrict__ img, unsigned short* __restrict__ P, long long total)
{
    long long i = (long long)blockIdx.x * 256 + threadIdx.x;
    if (i >= total) return;
    int col = (int)(i % DMODEL);
    long long row = i / DMODEL;
    int b = (int)(row / NPATCH), p = (int)(row % NPATCH);
    int ch = col >> 8, rem = col & 255, py = rem >> 4, px = rem & 15;
    int iy = (p / 14) * 16 + py, ix = (p % 14) * 16 + px;
    P[i] = f2bf(img[(((long long)(b * 3 + ch)) * 224 + iy) * 224 + ix]);
}

// x[b,t,:] = (t==0 ? cls : Xp[b*196+t-1]) + pos_emb[t]
__global__ void assemble_x(const float* __restrict__ Xp, const float* __restrict__ cls,
                           const float* __restrict__ pos, float* __restrict__ x, long long total)
{
    long long i = (long long)blockIdx.x * 256 + threadIdx.x;
    if (i >= total) return;
    int col = (int)(i % DMODEL);
    long long row = i / DMODEL;
    int b = (int)(row / NTOK), t = (int)(row % NTOK);
    float v = (t == 0) ? cls[col] : Xp[((long long)(b * NPATCH + t - 1)) * DMODEL + col];
    x[i] = v + pos[t * DMODEL + col];
}

__global__ void f32_to_bf16(const float* __restrict__ in, unsigned short* __restrict__ out, long long n)
{
    long long i = (long long)blockIdx.x * 256 + threadIdx.x;
    if (i < n) out[i] = f2bf(in[i]);
}

// qkv[M,2304] bf16 -> q,k [bh,208,64] and vT [bh,64,208] (zero-padded rows)
__global__ void scatter_qkv(const unsigned short* __restrict__ qkv,
                            unsigned short* __restrict__ q, unsigned short* __restrict__ k,
                            unsigned short* __restrict__ vT)
{
    long long i = (long long)blockIdx.x * 256 + threadIdx.x;
    const long long total = (long long)BATCH * NHEADS * NTOKP * DK;
    if (i >= total) return;
    int d = (int)(i % DK);
    int t = (int)((i / DK) % NTOKP);
    int bh = (int)(i / ((long long)DK * NTOKP));
    int b = bh / NHEADS, h = bh % NHEADS;
    unsigned short qv = 0, kv = 0, vv = 0;
    if (t < NTOK) {
        long long base = ((long long)(b * NTOK + t)) * DQKV + h * DK + d;
        qv = qkv[base]; kv = qkv[base + DMODEL]; vv = qkv[base + 2 * DMODEL];
    }
    long long o = (long long)bh * NTOKP * DK;
    q[o + (long long)t * DK + d] = qv;
    k[o + (long long)t * DK + d] = kv;
    vT[(long long)bh * DK * NTOKP + (long long)d * NTOKP + t] = vv;
}

// softmax over j<197 of S*0.125, bf16 out, padded region zeroed. One block per row.
__global__ __launch_bounds__(256) void softmax_row(const float* __restrict__ S, unsigned short* __restrict__ attn)
{
    __shared__ float red[256];
    long long row = blockIdx.x;
    int i = (int)(row % NTOKP);
    int j = threadIdx.x;
    const float* sr = S + row * NTOKP;
    bool valid = (i < NTOK) && (j < NTOK);
    float sv = valid ? sr[j] * 0.125f : -1e30f;
    red[j] = sv; __syncthreads();
    for (int s = 128; s > 0; s >>= 1) { if (j < s) red[j] = fmaxf(red[j], red[j + s]); __syncthreads(); }
    float mx = red[0]; __syncthreads();
    float e = valid ? expf(sv - mx) : 0.0f;
    float sum = blockReduceSum(e, red);
    if (j < NTOKP) {
        float o = (valid && sum > 0.0f) ? e / sum : 0.0f;
        attn[row * NTOKP + j] = f2bf(o);
    }
}

// final LN(cls row) -> logits, and logits_mean from y_mean cls row. One block per batch.
__global__ __launch_bounds__(256) void head_kernel(
    const float* __restrict__ x, const float* __restrict__ lnf_w, const float* __restrict__ lnf_b,
    const float* __restrict__ cls_w, const float* __restrict__ cls_b,
    const float* __restrict__ ymean, float* __restrict__ out)
{
    __shared__ float red[256];
    __shared__ float hrow[DMODEL];
    __shared__ float crow[DMODEL];
    int b = blockIdx.x, tid = threadIdx.x;
    const float* xr = x + (long long)(b * NTOK) * DMODEL;
    const float* yr = ymean + (long long)(b * NTOK) * DMODEL;
    float v[3]; float s = 0.0f;
    #pragma unroll
    for (int i = 0; i < 3; ++i) { v[i] = xr[tid + 256 * i]; s += v[i]; }
    float mu = blockReduceSum(s, red) * (1.0f / DMODEL);
    float vs = 0.0f;
    #pragma unroll
    for (int i = 0; i < 3; ++i) { float d = v[i] - mu; vs += d * d; }
    float var = blockReduceSum(vs, red) * (1.0f / DMODEL);
    float inv = rsqrtf(var + 1e-6f);
    #pragma unroll
    for (int i = 0; i < 3; ++i) {
        int c = tid + 256 * i;
        hrow[c] = (v[i] - mu) * inv * lnf_w[c] + lnf_b[c];
        crow[c] = yr[c];
    }
    __syncthreads();
    if (tid < 10) {
        float l1 = cls_b[tid], l2 = cls_b[tid];
        const float* wr = cls_w + tid * DMODEL;
        for (int c = 0; c < DMODEL; ++c) { float w = wr[c]; l1 += hrow[c] * w; l2 += crow[c] * w; }
        out[b * 10 + tid] = l1;
        out[BATCH * 10 + b * 10 + tid] = l2;
    }
}

// logits_std: var_vec[m] = sigma^2 * sum_o out_w[o,m]^2; std[n] = sqrt(sum_m var_vec[m]*cls_w[n,m]^2)
__global__ __launch_bounds__(256) void var_kernel(
    const float* __restrict__ out_w_last, const float* __restrict__ sigma11,
    const float* __restrict__ cls_w, float* __restrict__ out)
{
    __shared__ float vvec[DMODEL];
    int tid = threadIdx.x;
    float s2 = sigma11[0] * sigma11[0];
    for (int m = tid; m < DMODEL; m += 256) {
        float acc = 0.0f;
        for (int o = 0; o < DMODEL; ++o) { float w = out_w_last[(long long)o * DMODEL + m]; acc += w * w; }
        vvec[m] = acc * s2;
    }
    __syncthreads();
    if (tid < 10) {
        float a = 0.0f;
        const float* wr = cls_w + tid * DMODEL;
        for (int m = 0; m < DMODEL; ++m) { float w = wr[m]; a += vvec[m] * w * w; }
        float sd = sqrtf(a);
        for (int b = 0; b < BATCH; ++b) out[2 * BATCH * 10 + b * 10 + tid] = sd;
    }
}

// ---------------- host side ----------------
static inline size_t alignUp(size_t v, size_t a) { return (v + a - 1) & ~(a - 1); }

extern "C" void kernel_launch(void* const* d_in, const int* in_sizes, int n_in,
                              void* d_out, int out_size, void* d_ws, size_t ws_size,
                              hipStream_t stream)
{
    (void)in_sizes; (void)n_in; (void)out_size; (void)ws_size;
    const float* image    = (const float*)d_in[0];
    const float* conv_w   = (const float*)d_in[1];
    const float* conv_b   = (const float*)d_in[2];
    const float* cls_tok  = (const float*)d_in[3];
    const float* pos_emb  = (const float*)d_in[4];
    const float* ln1_w    = (const float*)d_in[5];
    const float* ln1_b    = (const float*)d_in[6];
    const float* qkv_w    = (const float*)d_in[7];
    const float* qkv_b    = (const float*)d_in[8];
    const float* out_w    = (const float*)d_in[9];
    const float* out_b    = (const float*)d_in[10];
    const float* ln2_w    = (const float*)d_in[11];
    const float* ln2_b    = (const float*)d_in[12];
    const float* mlp_w1   = (const float*)d_in[13];
    const float* mlp_b1   = (const float*)d_in[14];
    const float* mlp_w2   = (const float*)d_in[15];
    const float* mlp_b2   = (const float*)d_in[16];
    const float* sigmas   = (const float*)d_in[17];
    const float* lnf_w    = (const float*)d_in[18];
    const float* lnf_b    = (const float*)d_in[19];
    const float* cls_w    = (const float*)d_in[20];
    const float* cls_b    = (const float*)d_in[21];
    float* out = (float*)d_out;

    // workspace carve-out
    char* ws = (char*)d_ws;
    size_t cur = 0;
    auto carve = [&](size_t bytes) { size_t o = cur; cur = alignUp(cur + bytes, 256); return (void*)(ws + o); };

    unsigned short* convw_bf = (unsigned short*)carve((size_t)DMODEL * DMODEL * 2);
    unsigned short* qkvw_bf  = (unsigned short*)carve((size_t)NLAYERS * DQKV * DMODEL * 2);
    unsigned short* outw_bf  = (unsigned short*)carve((size_t)NLAYERS * DMODEL * DMODEL * 2);
    unsigned short* m1w_bf   = (unsigned short*)carve((size_t)NLAYERS * DFF * DMODEL * 2);
    unsigned short* m2w_bf   = (unsigned short*)carve((size_t)NLAYERS * DMODEL * DFF * 2);
    unsigned short* Pbf      = (unsigned short*)carve((size_t)MPATCH * DMODEL * 2);
    float*          Xp       = (float*)carve((size_t)MPATCH * DMODEL * 4);
    float*          x        = (float*)carve((size_t)MTOK * DMODEL * 4);
    unsigned short* hbf      = (unsigned short*)carve((size_t)MTOK * DMODEL * 2);
    unsigned short* qkvbf    = (unsigned short*)carve((size_t)MTOK * DQKV * 2);
    unsigned short* qb       = (unsigned short*)carve((size_t)BATCH * NHEADS * NTOKP * DK * 2);
    unsigned short* kb       = (unsigned short*)carve((size_t)BATCH * NHEADS * NTOKP * DK * 2);
    unsigned short* vT       = (unsigned short*)carve((size_t)BATCH * NHEADS * DK * NTOKP * 2);
    float*          Sbuf     = (float*)carve((size_t)BATCH * NHEADS * NTOKP * NTOKP * 4);
    unsigned short* attn     = (unsigned short*)carve((size_t)BATCH * NHEADS * NTOKP * NTOKP * 2);
    float*          zf32     = (float*)carve((size_t)MTOK * DMODEL * 4);
    unsigned short* zbf      = (unsigned short*)carve((size_t)MTOK * DMODEL * 2);
    unsigned short* h1bf     = (unsigned short*)carve((size_t)MTOK * DFF * 2);
    float*          ymbuf    = (float*)carve((size_t)MTOK * DMODEL * 4);

    auto cvt = [&](const float* src, unsigned short* dst, long long n) {
        f32_to_bf16<<<(unsigned)((n + 255) / 256), 256, 0, stream>>>(src, dst, n);
    };

    // ---- weight conversion (fp32 -> bf16), once per call ----
    cvt(conv_w, convw_bf, (long long)DMODEL * DMODEL);
    cvt(qkv_w,  qkvw_bf,  (long long)NLAYERS * DQKV * DMODEL);
    cvt(out_w,  outw_bf,  (long long)NLAYERS * DMODEL * DMODEL);
    cvt(mlp_w1, m1w_bf,   (long long)NLAYERS * DFF * DMODEL);
    cvt(mlp_w2, m2w_bf,   (long long)NLAYERS * DMODEL * DFF);

    // ---- patch embedding ----
    {
        long long tot = (long long)MPATCH * DMODEL;
        patch_gather<<<(unsigned)((tot + 255) / 256), 256, 0, stream>>>(image, Pbf, tot);
        dim3 g((DMODEL + BN - 1) / BN, (MPATCH + BM - 1) / BM, 1);
        gemm_bf16_wmma<EPI_F32><<<g, 256, 0, stream>>>(
            Pbf, DMODEL, 0, convw_bf, DMODEL, 0,
            Xp, DMODEL, 0, 0, 1, MPATCH, DMODEL, DMODEL, MPATCH,
            conv_b, nullptr, nullptr);
        long long tot2 = (long long)MTOK * DMODEL;
        assemble_x<<<(unsigned)((tot2 + 255) / 256), 256, 0, stream>>>(Xp, cls_tok, pos_emb, x, tot2);
    }

    const int BH = BATCH * NHEADS;
    for (int l = 0; l < NLAYERS; ++l) {
        // LN1 -> h (bf16)
        ln_kernel<<<MTOK, 256, 0, stream>>>(x, ln1_w + l * DMODEL, ln1_b + l * DMODEL, hbf);
        // QKV GEMM -> bf16 [MTOK, 2304]
        {
            dim3 g((DQKV + BN - 1) / BN, (MTOK + BM - 1) / BM, 1);
            gemm_bf16_wmma<EPI_BF16><<<g, 256, 0, stream>>>(
                hbf, DMODEL, 0, qkvw_bf + (size_t)l * DQKV * DMODEL, DMODEL, 0,
                qkvbf, DQKV, 0, 0, 1, MTOK, DQKV, DMODEL, MTOK,
                qkv_b + l * DQKV, nullptr, nullptr);
        }
        // scatter to padded q/k/vT
        {
            long long tot = (long long)BH * NTOKP * DK;
            scatter_qkv<<<(unsigned)((tot + 255) / 256), 256, 0, stream>>>(qkvbf, qb, kb, vT);
        }
        // scores = q @ k^T  (per bh)
        {
            dim3 g((NTOKP + BN - 1) / BN, (NTOKP + BM - 1) / BM, BH);
            gemm_bf16_wmma<EPI_F32><<<g, 256, 0, stream>>>(
                qb, DK, (long long)NTOKP * DK, kb, DK, (long long)NTOKP * DK,
                Sbuf, NTOKP, (long long)NTOKP * NTOKP, 0, 1,
                NTOKP, NTOKP, DK, NTOKP, nullptr, nullptr, nullptr);
        }
        // softmax rows
        softmax_row<<<BH * NTOKP, 256, 0, stream>>>(Sbuf, attn);
        // z = attn @ vT^T, scattered into [B,197,768] layout (head-concat)
        {
            dim3 g((DK + BN - 1) / BN, (NTOKP + BM - 1) / BM, BH);
            gemm_bf16_wmma<EPI_F32><<<g, 256, 0, stream>>>(
                attn, NTOKP, (long long)NTOKP * NTOKP, vT, NTOKP, (long long)DK * NTOKP,
                zf32, DMODEL, (long long)NTOK * DMODEL, DK, NHEADS,
                NTOKP, DK, NTOKP, NTOK, nullptr, nullptr, nullptr);
        }
        cvt(zf32, zbf, (long long)MTOK * DMODEL);
        // out-proj (+residual into x; last layer also captures y_mean)
        {
            dim3 g((DMODEL + BN - 1) / BN, (MTOK + BM - 1) / BM, 1);
            if (l == NLAYERS - 1)
                gemm_bf16_wmma<EPI_RES_YMEAN><<<g, 256, 0, stream>>>(
                    zbf, DMODEL, 0, outw_bf + (size_t)l * DMODEL * DMODEL, DMODEL, 0,
                    x, DMODEL, 0, 0, 1, MTOK, DMODEL, DMODEL, MTOK,
                    out_b + l * DMODEL, x, ymbuf);
            else
                gemm_bf16_wmma<EPI_RES><<<g, 256, 0, stream>>>(
                    zbf, DMODEL, 0, outw_bf + (size_t)l * DMODEL * DMODEL, DMODEL, 0,
                    x, DMODEL, 0, 0, 1, MTOK, DMODEL, DMODEL, MTOK,
                    out_b + l * DMODEL, x, nullptr);
        }
        // LN2 -> h
        ln_kernel<<<MTOK, 256, 0, stream>>>(x, ln2_w + l * DMODEL, ln2_b + l * DMODEL, hbf);
        // MLP1 + exact GELU -> bf16 [MTOK, 3072]
        {
            dim3 g((DFF + BN - 1) / BN, (MTOK + BM - 1) / BM, 1);
            gemm_bf16_wmma<EPI_GELU_BF16><<<g, 256, 0, stream>>>(
                hbf, DMODEL, 0, m1w_bf + (size_t)l * DFF * DMODEL, DMODEL, 0,
                h1bf, DFF, 0, 0, 1, MTOK, DFF, DMODEL, MTOK,
                mlp_b1 + l * DFF, nullptr, nullptr);
        }
        // MLP2 + residual -> x
        {
            dim3 g((DMODEL + BN - 1) / BN, (MTOK + BM - 1) / BM, 1);
            gemm_bf16_wmma<EPI_RES><<<g, 256, 0, stream>>>(
                h1bf, DFF, 0, m2w_bf + (size_t)l * DMODEL * DFF, DFF, 0,
                x, DMODEL, 0, 0, 1, MTOK, DMODEL, DFF, MTOK,
                mlp_b2 + l * DMODEL, x, nullptr);
        }
    }

    // ---- probabilistic head ----
    head_kernel<<<BATCH, 256, 0, stream>>>(x, lnf_w, lnf_b, cls_w, cls_b, ymbuf, out);
    var_kernel<<<1, 256, 0, stream>>>(out_w + (size_t)(NLAYERS - 1) * DMODEL * DMODEL,
                                      sigmas + (NLAYERS - 1), cls_w, out);
}
